// SCAEAttention_66520453481191
// MI455X (gfx1250) — matrix operand order repaired and checked by reading, not compiled, not run on hardware
//
#include <hip/hip_runtime.h>
#include <hip/hip_bf16.h>

typedef __attribute__((ext_vector_type(16))) __bf16 v16bf;
typedef __attribute__((ext_vector_type(8)))  float   v8f;

#define B_ 4
#define S_ 1024
#define H_ 8
#define D_ 512
#define F_ 2048
#define K_TOP 64

#define BM 128
#define BN 128
#define BK 32

// ---------------------------------------------------------------------------
// elementwise prep kernels
// ---------------------------------------------------------------------------
__global__ void k_inv(const float* __restrict__ ln, float* __restrict__ inv, int n) {
    int i = blockIdx.x * 256 + threadIdx.x;
    if (i < n) inv[i] = 1.0f / ln[i];
}

// out_bf16[i] = bf16(in[i] * inv[i >> shift])   (shift = log2(inner dim))
__global__ void k_scale_cvt(const float* __restrict__ in, const float* __restrict__ inv,
                            __bf16* __restrict__ out, int shift, int n) {
    int i = blockIdx.x * 256 + threadIdx.x;
    if (i < n) out[i] = (__bf16)(in[i] * inv[i >> shift]);
}

__global__ void k_cvt(const float* __restrict__ in, __bf16* __restrict__ out, int n) {
    int i = blockIdx.x * 256 + threadIdx.x;
    if (i < n) out[i] = (__bf16)in[i];
}

// ---------------------------------------------------------------------------
// async global->LDS copy of 32 contiguous bytes (two B128 transfers).
// dst: generic pointer into __shared__ (low 32 bits == LDS offset on gfx1250)
// src: global pointer. Tracked with ASYNCcnt; no VGPR staging.
// ---------------------------------------------------------------------------
__device__ __forceinline__ void async_copy_b256(void* dst_lds, const void* src_glb) {
    unsigned           lds_off = (unsigned)(uintptr_t)dst_lds;
    unsigned long long gaddr   = (unsigned long long)(uintptr_t)src_glb;
    asm volatile(
        "global_load_async_to_lds_b128 %0, %1, off\n\t"
        "global_load_async_to_lds_b128 %0, %1, off offset:16"
        :: "v"(lds_off), "v"(gaddr) : "memory");
}

__device__ __forceinline__ void async_wait0() {
    asm volatile("s_wait_asynccnt 0x0" ::: "memory");
}

// ---------------------------------------------------------------------------
// generic tiled bf16 GEMM with fp32 WMMA accumulation
//   C(M,N) = A(MxK) * B   (B is KxN row-major, or NxK row-major if transB)
//   epilogue: optional accumulate into fp32 C, optional fp32 mask multiply,
//             output fp32 or bf16.
//   grid = (N/BN, M/BM), block = 256 threads (8 waves, wave32)
// ---------------------------------------------------------------------------
__global__ __launch_bounds__(256)
void k_gemm(const __bf16* __restrict__ A, int lda,
            const __bf16* __restrict__ Bm, int ldb, int transB,
            void* __restrict__ Cp, int ldc,
            int K, int accum, int out_bf16,
            const float* __restrict__ mask, int ldmask)
{
    __shared__ __bf16 sA[BM][BK];   // row-major (m, k)
    __shared__ __bf16 sB[BN][BK];   // n-major  (n, k) -> contiguous K per column

    const int tid  = threadIdx.x;
    const int lane = tid & 31;
    const int wave = tid >> 5;           // 0..7
    const int wm   = wave >> 1;          // 0..3 : 32-row group
    const int wn   = wave & 1;           // 0..1 : 64-col group
    const int hh   = lane >> 4;          // half-wave select
    const int l16  = lane & 15;

    const int bm = blockIdx.y * BM;
    const int bn = blockIdx.x * BN;

    v8f acc[2][4];
#pragma unroll
    for (int tm = 0; tm < 2; ++tm)
#pragma unroll
        for (int tn = 0; tn < 4; ++tn)
#pragma unroll
            for (int e = 0; e < 8; ++e) acc[tm][tn][e] = 0.0f;

    const int arow   = tid >> 1;   // 0..127
    const int achunk = tid & 1;    // 0..1  (32 bytes each half-row)

    for (int k0 = 0; k0 < K; k0 += BK) {
        // ---- A tile: async global->LDS (no VGPR staging, ASYNCcnt) ----
        async_copy_b256(&sA[arow][achunk * 16],
                        A + (size_t)(bm + arow) * lda + k0 + achunk * 16);
        if (k0 + BK < K)  // gfx1250 global_prefetch_b8 for next A tile
            __builtin_prefetch(A + (size_t)(bm + arow) * lda + k0 + BK, 0, 1);

        // ---- B tile into n-major LDS layout ----
        if (transB) {
            // B is (N x K) row-major: contiguous copy -> async-to-LDS
            async_copy_b256(&sB[arow][achunk * 16],
                            Bm + (size_t)(bn + arow) * ldb + k0 + achunk * 16);
        } else {
            // B is (K x N) row-major: transpose on the fly via VGPRs
            const int kk = tid >> 3;          // 0..31
            const int nb = (tid & 7) * 8;     // uint32 index base along n
            const unsigned int* g = (const unsigned int*)(Bm + (size_t)(k0 + kk) * ldb + bn);
#pragma unroll
            for (int j = 0; j < 8; ++j) {
                union { unsigned int u; __bf16 h[2]; } cv;
                cv.u = g[nb + j];
                const int n = (nb + j) * 2;
                sB[n][kk]     = cv.h[0];
                sB[n + 1][kk] = cv.h[1];
            }
        }
        async_wait0();            // this wave's async LDS writes are visible
        __syncthreads();          // all waves' tiles are in LDS

        // ---- fragments per ISA 16-bit A(16x32)/B(32x16) wave32 layouts ----
        union Frag { v16bf v; uint4 q[2]; };
        Frag af[2], bfg[4];
#pragma unroll
        for (int tm = 0; tm < 2; ++tm) {
            const int r = wm * 32 + tm * 16 + l16;
            af[tm].q[0] = *(const uint4*)(&sA[r][hh * 8]);
            af[tm].q[1] = *(const uint4*)(&sA[r][16 + hh * 8]);
        }
#pragma unroll
        for (int tn = 0; tn < 4; ++tn) {
            const int c = wn * 64 + tn * 16 + l16;
            bfg[tn].q[0] = *(const uint4*)(&sB[c][hh * 8]);
            bfg[tn].q[1] = *(const uint4*)(&sB[c][16 + hh * 8]);
        }

#pragma unroll
        for (int tm = 0; tm < 2; ++tm)
#pragma unroll
            for (int tn = 0; tn < 4; ++tn)
                acc[tm][tn] = __builtin_amdgcn_wmma_f32_16x16x32_bf16(
                    false, af[tm].v, false, bfg[tn].v,
                    (short)0, acc[tm][tn], false, false);

        __syncthreads();
    }

    // ---- epilogue (C/D layout: VGPR j -> row j + hh*8, col l16) ----
    float*  Cf = (float*)Cp;
    __bf16* Cb = (__bf16*)Cp;
#pragma unroll
    for (int tm = 0; tm < 2; ++tm) {
#pragma unroll
        for (int tn = 0; tn < 4; ++tn) {
#pragma unroll
            for (int j = 0; j < 8; ++j) {
                const int r = bm + wm * 32 + tm * 16 + hh * 8 + j;
                const int c = bn + wn * 64 + tn * 16 + l16;
                float v = acc[tm][tn][j];
                const size_t idx = (size_t)r * ldc + c;
                if (accum) v += Cf[idx];
                if (mask)  v *= mask[(size_t)r * ldmask + c];
                if (out_bf16) Cb[idx] = (__bf16)v;
                else          Cf[idx] = v;
            }
        }
    }
}

// ---------------------------------------------------------------------------
// bias helpers
// ---------------------------------------------------------------------------
// w2[o] = sum_{h,i} W_OV[h,i,o] * up_b_dec[i]
__global__ void k_w2(const float* __restrict__ WOV, const float* __restrict__ upb,
                     float* __restrict__ w2) {
    int o = blockIdx.x * 256 + threadIdx.x;
    if (o < D_) {
        float acc = 0.0f;
        for (int h = 0; h < H_; ++h)
            for (int i = 0; i < D_; ++i)
                acc += WOV[((size_t)h * D_ + i) * D_ + o] * upb[i];
        w2[o] = acc;
    }
}

// biasf[f]  = down_enc[f,:] . (b_O - b_dec) + enc_bias[f]
// bias2f[f] = down_enc[f,:] . w2
__global__ __launch_bounds__(256)
void k_biasf(const float* __restrict__ down, const float* __restrict__ bO,
             const float* __restrict__ bdec, const float* __restrict__ encb,
             const float* __restrict__ w2,
             float* __restrict__ biasf, float* __restrict__ bias2f) {
    __shared__ float s1[256], s2[256];
    const int f = blockIdx.x;
    float a = 0.0f, b = 0.0f;
    for (int o = threadIdx.x; o < D_; o += 256) {
        const float de = down[(size_t)f * D_ + o];
        a += de * (bO[o] - bdec[o]);
        b += de * w2[o];
    }
    s1[threadIdx.x] = a; s2[threadIdx.x] = b;
    __syncthreads();
    for (int s = 128; s > 0; s >>= 1) {
        if (threadIdx.x < s) { s1[threadIdx.x] += s1[threadIdx.x + s]; s2[threadIdx.x] += s2[threadIdx.x + s]; }
        __syncthreads();
    }
    if (threadIdx.x == 0) { biasf[f] = s1[0] + encb[f]; bias2f[f] = s2[0]; }
}

// approx[b,s,f] += biasf[f] + inv[b,s] * bias2f[f]
__global__ void k_addbias(float* __restrict__ approx, const float* __restrict__ biasf,
                          const float* __restrict__ bias2f, const float* __restrict__ inv, int n) {
    int i = blockIdx.x * 256 + threadIdx.x;
    if (i < n) {
        const int f  = i & (F_ - 1);
        const int bs = i >> 11;
        approx[i] += biasf[f] + inv[bs] * bias2f[f];
    }
}

// ---------------------------------------------------------------------------
// fused top-k + relu + sparse decode:  recon[b,s,:] = b_dec + sum_k v_k*dec_w[:,idx_k]
// one block per (b,s) row
// ---------------------------------------------------------------------------
__global__ __launch_bounds__(256)
void k_topk_recon(const float* __restrict__ approx, const float* __restrict__ decw,
                  const float* __restrict__ bdec, float* __restrict__ out) {
    __shared__ float vals[F_];
    __shared__ float sval[256];
    __shared__ int   sidx[256];
    __shared__ float topv[K_TOP];
    __shared__ int   topi[K_TOP];

    const int bs = blockIdx.x;
    const float* row = approx + (size_t)bs * F_;
    for (int f = threadIdx.x; f < F_; f += 256) vals[f] = row[f];
    __syncthreads();

    for (int it = 0; it < K_TOP; ++it) {
        float best = -INFINITY; int bi = -1;
        for (int f = threadIdx.x; f < F_; f += 256) {
            const float v = vals[f];
            if (v > best) { best = v; bi = f; }
        }
        sval[threadIdx.x] = best; sidx[threadIdx.x] = bi;
        __syncthreads();
        for (int s = 128; s > 0; s >>= 1) {
            if (threadIdx.x < s && sval[threadIdx.x + s] > sval[threadIdx.x]) {
                sval[threadIdx.x] = sval[threadIdx.x + s];
                sidx[threadIdx.x] = sidx[threadIdx.x + s];
            }
            __syncthreads();
        }
        if (threadIdx.x == 0) {
            topv[it] = sval[0];
            topi[it] = sidx[0];
            vals[sidx[0]] = -INFINITY;
        }
        __syncthreads();
    }

    for (int d = threadIdx.x; d < D_; d += 256) {
        float acc = bdec[d];
#pragma unroll 8
        for (int j = 0; j < K_TOP; ++j) {
            float v = topv[j];
            v = v > 0.0f ? v : 0.0f;
            acc += v * decw[(size_t)d * F_ + topi[j]];
        }
        out[(size_t)bs * D_ + d] = acc;
    }
}

// ---------------------------------------------------------------------------
// host orchestration
// ---------------------------------------------------------------------------
extern "C" void kernel_launch(void* const* d_in, const int* in_sizes, int n_in,
                              void* d_out, int out_size, void* d_ws, size_t ws_size,
                              hipStream_t stream)
{
    const float* embed  = (const float*)d_in[0];
    const float* ln1    = (const float*)d_in[1];
    const float* probs  = (const float*)d_in[2];
    const float* pruned = (const float*)d_in[3];
    // d_in[4] feature_buffer: unused (decode fused)
    const float* WOV    = (const float*)d_in[5];
    const float* down   = (const float*)d_in[6];
    const float* encb   = (const float*)d_in[7];
    const float* decw   = (const float*)d_in[8];
    const float* bdec   = (const float*)d_in[9];
    const float* bO     = (const float*)d_in[10];
    const float* updec  = (const float*)d_in[11];
    const float* upb    = (const float*)d_in[12];
    const float* cmask  = (const float*)d_in[13];
    float* recon = (float*)d_out;

    char* ws = (char*)d_ws;
    size_t off = 0;
    auto alloc = [&](size_t bytes) -> void* {
        void* p = ws + off;
        off = (off + bytes + 255) & ~(size_t)255;
        return p;
    };

    float*  inv    = (float*) alloc(sizeof(float) * B_ * S_);
    __bf16* xb     = (__bf16*)alloc(2ull * B_ * S_ * D_);
    __bf16* factsb = (__bf16*)alloc(2ull * B_ * S_ * F_);
    __bf16* wovb   = (__bf16*)alloc(2ull * H_ * D_ * D_);
    __bf16* downb  = (__bf16*)alloc(2ull * F_ * D_);
    __bf16* upbf   = (__bf16*)alloc(2ull * D_ * F_);
    __bf16* probsb = (__bf16*)alloc(2ull * B_ * H_ * S_ * S_);
    __bf16* t1b    = (__bf16*)alloc(2ull * B_ * H_ * S_ * D_);
    float*  movedf = (float*) alloc(4ull * B_ * S_ * D_);
    __bf16* movedb = (__bf16*)alloc(2ull * B_ * S_ * D_);
    __bf16* tmpb   = (__bf16*)alloc(2ull * H_ * F_ * D_);
    __bf16* vwb    = (__bf16*)alloc(2ull * H_ * F_ * F_);
    __bf16* zb     = (__bf16*)alloc(2ull * B_ * H_ * S_ * F_);
    float*  approx = (float*) alloc(4ull * B_ * S_ * F_);
    float*  w2     = (float*) alloc(sizeof(float) * D_);
    float*  biasf  = (float*) alloc(sizeof(float) * F_);
    float*  bias2f = (float*) alloc(sizeof(float) * F_);

    const int nBS = B_ * S_;

    // --- prep / casts ---
    k_inv<<<(nBS + 255) / 256, 256, 0, stream>>>(ln1, inv, nBS);
    k_scale_cvt<<<(B_*S_*D_) / 256, 256, 0, stream>>>(embed,  inv, xb,     9,  B_*S_*D_);
    k_scale_cvt<<<(B_*S_*F_) / 256, 256, 0, stream>>>(pruned, inv, factsb, 11, B_*S_*F_);
    k_cvt<<<(H_*D_*D_) / 256, 256, 0, stream>>>(WOV,   wovb,  H_*D_*D_);
    k_cvt<<<(F_*D_)    / 256, 256, 0, stream>>>(down,  downb, F_*D_);
    k_cvt<<<(D_*F_)    / 256, 256, 0, stream>>>(updec, upbf,  D_*F_);
    k_cvt<<<(B_*H_*S_*S_) / 256, 256, 0, stream>>>(probs, probsb, B_*H_*S_*S_);

    // --- t1[b,h] = x[b] (SxD) @ W_OV[h] (DxD) -> bf16 ---
    for (int b = 0; b < B_; ++b)
        for (int h = 0; h < H_; ++h)
            k_gemm<<<dim3(D_/BN, S_/BM), 256, 0, stream>>>(
                xb + (size_t)b*S_*D_, D_, wovb + (size_t)h*D_*D_, D_, 0,
                t1b + ((size_t)b*H_ + h)*S_*D_, D_, D_, 0, 1, nullptr, 0);

    // --- moved[b] = sum_h probs[b,h] (SxS) @ t1[b,h] (SxD) -> fp32 accumulate ---
    for (int b = 0; b < B_; ++b)
        for (int h = 0; h < H_; ++h)
            k_gemm<<<dim3(D_/BN, S_/BM), 256, 0, stream>>>(
                probsb + ((size_t)b*H_ + h)*S_*S_, S_,
                t1b + ((size_t)b*H_ + h)*S_*D_, D_, 0,
                movedf + (size_t)b*S_*D_, D_, S_, (h > 0) ? 1 : 0, 0, nullptr, 0);
    k_cvt<<<(B_*S_*D_) / 256, 256, 0, stream>>>(movedf, movedb, B_*S_*D_);

    // --- approx1[b] = moved[b] (SxD) @ down_enc^T (DxF) -> fp32 (overwrite) ---
    for (int b = 0; b < B_; ++b)
        k_gemm<<<dim3(F_/BN, S_/BM), 256, 0, stream>>>(
            movedb + (size_t)b*S_*D_, D_, downb, D_, 1,
            approx + (size_t)b*S_*F_, F_, D_, 0, 0, nullptr, 0);

    // --- tmp[h] = down_enc (FxD) @ W_OV[h]^T (DxD) -> bf16 ---
    for (int h = 0; h < H_; ++h)
        k_gemm<<<dim3(D_/BN, F_/BM), 256, 0, stream>>>(
            downb, D_, wovb + (size_t)h*D_*D_, D_, 1,
            tmpb + (size_t)h*F_*D_, D_, D_, 0, 1, nullptr, 0);

    // --- vw[h] = (tmp[h] (FxD) @ up_dec (DxF)) * conn_mask -> bf16 ---
    for (int h = 0; h < H_; ++h)
        k_gemm<<<dim3(F_/BN, F_/BM), 256, 0, stream>>>(
            tmpb + (size_t)h*F_*D_, D_, upbf, F_, 0,
            vwb + (size_t)h*F_*F_, F_, D_, 0, 1, cmask, F_);

    // --- z[b,h] = probs[b,h] (SxS) @ facts[b] (SxF) -> bf16 ---
    for (int b = 0; b < B_; ++b)
        for (int h = 0; h < H_; ++h)
            k_gemm<<<dim3(F_/BN, S_/BM), 256, 0, stream>>>(
                probsb + ((size_t)b*H_ + h)*S_*S_, S_,
                factsb + (size_t)b*S_*F_, F_, 0,
                zb + ((size_t)b*H_ + h)*S_*F_, F_, S_, 0, 1, nullptr, 0);

    // --- approx[b] += sum_h z[b,h] (SxF) @ vw[h]^T (FxF) -> fp32 accumulate ---
    for (int b = 0; b < B_; ++b)
        for (int h = 0; h < H_; ++h)
            k_gemm<<<dim3(F_/BN, S_/BM), 256, 0, stream>>>(
                zb + ((size_t)b*H_ + h)*S_*F_, F_,
                vwb + (size_t)h*F_*F_, F_, 1,
                approx + (size_t)b*S_*F_, F_, F_, 1, 0, nullptr, 0);

    // --- bias folding ---
    k_w2<<<(D_ + 255) / 256, 256, 0, stream>>>(WOV, upb, w2);
    k_biasf<<<F_, 256, 0, stream>>>(down, bO, bdec, encb, w2, biasf, bias2f);
    k_addbias<<<(B_*S_*F_) / 256, 256, 0, stream>>>(approx, biasf, bias2f, inv, B_*S_*F_);

    // --- fused top-k + relu + sparse decode ---
    k_topk_recon<<<nBS, 256, 0, stream>>>(approx, decw, bdec, recon);
}
